// SparseEventNet_12335146074172
// MI455X (gfx1250) — matrix operand order repaired and checked by reading, not compiled, not run on hardware
//
#include <hip/hip_runtime.h>
#include <hip/hip_bf16.h>
#include <stdint.h>
#include <stddef.h>

// ---------------------------------------------------------------------------
// Types / constants
// ---------------------------------------------------------------------------
typedef __attribute__((ext_vector_type(4)))  float    f32x4;
typedef __attribute__((ext_vector_type(2)))  float    v2f;
typedef __attribute__((ext_vector_type(8)))  float    v8f;
typedef __attribute__((ext_vector_type(16))) _Float16 v16h;

#define C_IN  128   // channels from sparse conv stack
#define D_IN  256   // 2*C_IN (mean || max)
#define H1    128
#define H2    64
#define H3    32
#define NCLS  4

// fp32 WMMA (16x16x4) preferred for precision; fall back to the
// codegen-confirmed f16 (16x16x32) path if the builtin is not declared.
#if defined(__has_builtin)
#if __has_builtin(__builtin_amdgcn_wmma_f32_16x16x4_f32)
#define USE_WMMA_F32 1
#endif
#endif
#ifndef USE_WMMA_F32
#define USE_WMMA_F32 0
#endif

// ---------------------------------------------------------------------------
// Kernel 0: init accumulators (seg_sum = 0, seg_max = -inf bits, counts = 0)
// ---------------------------------------------------------------------------
__global__ void sen_init_kernel(float* __restrict__ seg_sum,
                                unsigned* __restrict__ seg_max,
                                float* __restrict__ counts, int B) {
  int i = blockIdx.x * blockDim.x + threadIdx.x;
  int n = B * C_IN;
  if (i < n) {
    seg_sum[i] = 0.0f;
    seg_max[i] = 0xFF800000u;  // bit pattern of -inf
  }
  if (i < B) counts[i] = 0.0f;
}

// ---------------------------------------------------------------------------
// Kernel 1: streaming segment sum+max (bandwidth-bound part, ~44us floor)
// ---------------------------------------------------------------------------
__device__ inline void sen_atomic_max_f32(unsigned* addr, float v) {
  int iv = __float_as_int(v);
  if (iv >= 0) atomicMax((int*)addr, iv);
  else         atomicMin(addr, (unsigned)iv);
}

__device__ inline void sen_flush(float* __restrict__ seg_sum,
                                 unsigned* __restrict__ seg_max,
                                 float* __restrict__ counts,
                                 int bid, int c0, int lane,
                                 f32x4 s, f32x4 m, float cnt) {
  if (cnt <= 0.0f) return;
  float* ps = seg_sum + (size_t)bid * C_IN + c0;
  atomicAdd(ps + 0, s.x);
  atomicAdd(ps + 1, s.y);
  atomicAdd(ps + 2, s.z);
  atomicAdd(ps + 3, s.w);
  unsigned* pm = seg_max + (size_t)bid * C_IN + c0;
  sen_atomic_max_f32(pm + 0, m.x);
  sen_atomic_max_f32(pm + 1, m.y);
  sen_atomic_max_f32(pm + 2, m.z);
  sen_atomic_max_f32(pm + 3, m.w);
  if (lane == 0) atomicAdd(counts + bid, cnt);
}

__global__ __launch_bounds__(256)
void sen_segpool_kernel(const float* __restrict__ feat,
                        const int* __restrict__ bids,
                        float* __restrict__ seg_sum,
                        unsigned* __restrict__ seg_max,
                        float* __restrict__ counts,
                        int n_rows, int rows_per_wave) {
  const int lane = threadIdx.x & 31;
  const int wave = blockIdx.x * (blockDim.x >> 5) + (threadIdx.x >> 5);
  long r0 = (long)wave * rows_per_wave;
  if (r0 >= n_rows) return;
  long r1 = r0 + rows_per_wave;
  if (r1 > n_rows) r1 = n_rows;

  const int c0 = lane << 2;           // each lane owns 4 channels (32*4 = 128)
  const float NEG_INF = -__builtin_inff();
  f32x4 sum = {0.0f, 0.0f, 0.0f, 0.0f};
  f32x4 mx  = {NEG_INF, NEG_INF, NEG_INF, NEG_INF};
  float cnt = 0.0f;
  int cur = bids[r0];                 // sorted ids -> register accumulation

  for (long r = r0; r < r1; ++r) {
    int b = bids[r];                  // lane-uniform
    if (b != cur) {                   // segment boundary: rare flush
      sen_flush(seg_sum, seg_max, counts, cur, c0, lane, sum, mx, cnt);
      sum = (f32x4){0.0f, 0.0f, 0.0f, 0.0f};
      mx  = (f32x4){NEG_INF, NEG_INF, NEG_INF, NEG_INF};
      cnt = 0.0f;
      cur = b;
    }
    if (r + 8 < r1)                   // look-ahead: global_prefetch_b8
      __builtin_prefetch(feat + (size_t)(r + 8) * C_IN + c0, 0, 0);
    // streaming 1 GB >> 192 MB L2: non-temporal b128 load
    f32x4 v = __builtin_nontemporal_load(
        (const f32x4*)(feat + (size_t)r * C_IN + c0));
    sum += v;
    mx.x = fmaxf(mx.x, v.x);
    mx.y = fmaxf(mx.y, v.y);
    mx.z = fmaxf(mx.z, v.z);
    mx.w = fmaxf(mx.w, v.w);
    cnt += 1.0f;
  }
  sen_flush(seg_sum, seg_max, counts, cur, c0, lane, sum, mx, cnt);
}

// ---------------------------------------------------------------------------
// Kernel 2: WMMA MLP head. One wave per 16-row M-tile of the batch.
// C/D layout (ISA 7.12.2): lane = {l2 = col, hi}; acc[r] = D[r + 8*hi][l2].
// ---------------------------------------------------------------------------
__device__ inline v8f sen_tile_mma(const float* __restrict__ A, int lda,
                                   const float* __restrict__ Bm, int ldb,
                                   int K, v8f acc) {
  const int lane = threadIdx.x & 31;
  const int l2 = lane & 15;
  const int hi = lane >> 4;
#if USE_WMMA_F32
  // A 16x4 f32: lane l2 = row M; vgpr j -> K = k0 + j + 2*hi
  // B 4x16 f32: lane l2 = col N; vgpr j -> K = k0 + j + 2*hi
  for (int k0 = 0; k0 < K; k0 += 4) {
    v2f a, b;
    a[0] = A[l2 * lda + k0 + 2 * hi + 0];
    a[1] = A[l2 * lda + k0 + 2 * hi + 1];
    b[0] = Bm[(size_t)(k0 + 2 * hi + 0) * ldb + l2];
    b[1] = Bm[(size_t)(k0 + 2 * hi + 1) * ldb + l2];
    acc = __builtin_amdgcn_wmma_f32_16x16x4_f32(
        false, a, false, b, (short)0, acc, false, false);
  }
#else
  // Confirmed fallback: f32 accumulate over f16 inputs, K-step 32.
  for (int k0 = 0; k0 < K; k0 += 32) {
    v16h a, b;
#pragma unroll
    for (int i = 0; i < 16; ++i) {
      int vg = i >> 1, p = i & 1;
      int ka = 2 * vg + p + 8 * hi + ((vg >= 4) ? 8 : 0);  // ISA A 16x32 map
      int kb = i + 8 * hi + ((i >= 8) ? 8 : 0);             // B: same K map as A
      a[i] = (_Float16)A[l2 * lda + k0 + ka];
      b[i] = (_Float16)Bm[(size_t)(k0 + kb) * ldb + l2];
    }
    acc = __builtin_amdgcn_wmma_f32_16x16x32_f16(
        false, a, false, b, (short)0, acc, false, false);
  }
#endif
  return acc;
}

__global__ __launch_bounds__(32)
void sen_mlp_kernel(const float* __restrict__ seg_sum,
                    const unsigned* __restrict__ seg_max,
                    const float* __restrict__ counts,
                    const float* __restrict__ W1, const float* __restrict__ b1,
                    const float* __restrict__ W2, const float* __restrict__ b2,
                    const float* __restrict__ W3, const float* __restrict__ b3,
                    const float* __restrict__ W4, const float* __restrict__ b4,
                    float* __restrict__ out, int B) {
  __shared__ float sPool[16 * D_IN];  // 16 KB
  __shared__ float sA1[16 * H1];      // 8 KB
  __shared__ float sA2[16 * H2];      // 4 KB
  __shared__ float sA3[16 * H3];      // 2 KB

  const int tid = threadIdx.x;
  const int l2 = tid & 15;
  const int hi = tid >> 4;
  const int m0 = blockIdx.x * 16;

  // Stage A: pooled tile = [mean || max] for rows m0..m0+15
  for (int idx = tid; idx < 16 * D_IN; idx += 32) {
    int r = idx >> 8;           // D_IN == 256
    int c = idx & (D_IN - 1);
    int b = m0 + r;
    float v = 0.0f;
    if (b < B) {
      float cnt = counts[b];
      if (cnt > 0.0f) {
        v = (c < C_IN)
                ? seg_sum[(size_t)b * C_IN + c] / cnt
                : __uint_as_float(seg_max[(size_t)b * C_IN + (c - C_IN)]);
      }
    }
    sPool[idx] = v;
  }
  __syncthreads();

  // Layer 1: [16,256] @ [256,128] + b1, ReLU
  for (int nt = 0; nt < H1 / 16; ++nt) {
    v8f acc = {0.f, 0.f, 0.f, 0.f, 0.f, 0.f, 0.f, 0.f};
    acc = sen_tile_mma(sPool, D_IN, W1 + nt * 16, H1, D_IN, acc);
    int n = nt * 16 + l2;
    float bias = b1[n];
#pragma unroll
    for (int r = 0; r < 8; ++r) {
      float v = acc[r] + bias;
      sA1[(r + 8 * hi) * H1 + n] = v > 0.0f ? v : 0.0f;
    }
  }
  __syncthreads();

  // Layer 2: [16,128] @ [128,64] + b2, ReLU
  for (int nt = 0; nt < H2 / 16; ++nt) {
    v8f acc = {0.f, 0.f, 0.f, 0.f, 0.f, 0.f, 0.f, 0.f};
    acc = sen_tile_mma(sA1, H1, W2 + nt * 16, H2, H1, acc);
    int n = nt * 16 + l2;
    float bias = b2[n];
#pragma unroll
    for (int r = 0; r < 8; ++r) {
      float v = acc[r] + bias;
      sA2[(r + 8 * hi) * H2 + n] = v > 0.0f ? v : 0.0f;
    }
  }
  __syncthreads();

  // Layer 3: [16,64] @ [64,32] + b3, ReLU
  for (int nt = 0; nt < H3 / 16; ++nt) {
    v8f acc = {0.f, 0.f, 0.f, 0.f, 0.f, 0.f, 0.f, 0.f};
    acc = sen_tile_mma(sA2, H2, W3 + nt * 16, H3, H2, acc);
    int n = nt * 16 + l2;
    float bias = b3[n];
#pragma unroll
    for (int r = 0; r < 8; ++r) {
      float v = acc[r] + bias;
      sA3[(r + 8 * hi) * H3 + n] = v > 0.0f ? v : 0.0f;
    }
  }
  __syncthreads();

  // Layer 4: [16,32] @ [32,4] + b4  (N=4 < tile -> scalar, negligible)
  for (int idx = tid; idx < 16 * NCLS; idx += 32) {
    int r = idx >> 2;
    int j = idx & 3;
    int b = m0 + r;
    if (b < B) {
      float s = b4[j];
#pragma unroll
      for (int k = 0; k < H3; ++k) s += sA3[r * H3 + k] * W4[k * NCLS + j];
      out[(size_t)b * NCLS + j] = s;
    }
  }
}

// ---------------------------------------------------------------------------
// Host launcher
// ---------------------------------------------------------------------------
extern "C" void kernel_launch(void* const* d_in, const int* in_sizes, int n_in,
                              void* d_out, int out_size, void* d_ws,
                              size_t ws_size, hipStream_t stream) {
  const float* feat = (const float*)d_in[0];
  const int*   bids = (const int*)d_in[1];
  // d_in[2] = batch_size (device scalar, unused: B derived from out_size)
  const float* W1 = (const float*)d_in[3];
  const float* b1 = (const float*)d_in[4];
  const float* W2 = (const float*)d_in[5];
  const float* b2 = (const float*)d_in[6];
  const float* W3 = (const float*)d_in[7];
  const float* b3 = (const float*)d_in[8];
  const float* W4 = (const float*)d_in[9];
  const float* b4 = (const float*)d_in[10];

  const int N = in_sizes[0] / C_IN;   // active voxels
  const int B = out_size / NCLS;      // batch size (256)

  float*    seg_sum = (float*)d_ws;
  unsigned* seg_max =
      (unsigned*)((char*)d_ws + (size_t)B * C_IN * sizeof(float));
  float* counts =
      (float*)((char*)d_ws + 2 * (size_t)B * C_IN * sizeof(float));

  // 0) init accumulators
  {
    int n = B * C_IN;
    int blk = (n + 255) / 256;
    sen_init_kernel<<<blk, 256, 0, stream>>>(seg_sum, seg_max, counts, B);
  }

  // 1) streaming segment reduce: 8192 waves, contiguous row slices
  {
    const int TOTAL_WAVES = 8192;
    int rpw = (N + TOTAL_WAVES - 1) / TOTAL_WAVES;
    if (rpw < 1) rpw = 1;
    int blocks = TOTAL_WAVES / 8;  // 256 threads = 8 wave32 per block
    sen_segpool_kernel<<<blocks, 256, 0, stream>>>(feat, bids, seg_sum,
                                                   seg_max, counts, N, rpw);
  }

  // 2) WMMA MLP head: one wave per 16-row tile
  {
    int blocks = (B + 15) / 16;
    sen_mlp_kernel<<<blocks, 32, 0, stream>>>(seg_sum, seg_max, counts, W1, b1,
                                              W2, b2, W3, b3, W4, b4,
                                              (float*)d_out, B);
  }
}
